// LSTM_47261820125696
// MI455X (gfx1250) — matrix operand order repaired and checked by reading, not compiled, not run on hardware
//
#include <hip/hip_runtime.h>
#include <hip/hip_bf16.h>
#include <math.h>

// ---------------------------------------------------------------------------
// LSTM (S=16384, I=64, H=1024) + FC(1024->128->1), batch=1, only x[-1] needed.
//
// K1: xg = X @ W_ih.T + (b_ih+b_hh)   -- WMMA f16 GEMM, fp32 out (256MB ws)
// K2: persistent recurrence, 32 WGs, W_hh slab (128x1024 f16 = 256KB) in LDS,
//     4-chain WMMA GEMV per step, async global->LDS staging of h and xg
//     (overlapped with the GEMV), device-wide atomic barrier per step,
//     h double-buffered in global.
// K3: FC epilogue on final h only.
// ---------------------------------------------------------------------------

typedef __attribute__((ext_vector_type(16))) _Float16 v16h;
typedef __attribute__((ext_vector_type(8)))  _Float16 v8h;
typedef __attribute__((ext_vector_type(8)))  float    v8f;

#define S_LEN   16384
#define I_DIM   64
#define H_DIM   1024
#define G_DIM   4096       // 4*H
#define FC_DIM  128
#define NWG     32         // persistent workgroups (each: 256KB W_hh slab in LDS)
#define ROWS_PER_WG 128    // 4 gates * 32 hidden
#define HID_PER_WG  32

__device__ __forceinline__ v16h cat8(v8h lo, v8h hi) {
  return __builtin_shufflevector(lo, hi, 0,1,2,3,4,5,6,7,8,9,10,11,12,13,14,15);
}

__device__ __forceinline__ float sigmoidf(float x) {
  return 1.0f / (1.0f + __expf(-x));
}

// Low 32 bits of a shared-aperture flat address == LDS byte address (ISA 10.2).
__device__ __forceinline__ unsigned lds_u32(const void* p) {
  return (unsigned)(uintptr_t)p;
}

// --------------------------- K0: init state --------------------------------
__global__ void k0_init(_Float16* __restrict__ h_bf /*2*H doublebuf*/,
                        float* __restrict__ h_f32,
                        unsigned* __restrict__ cnt) {
  const int t = threadIdx.x;                 // 1024 threads
  h_bf[t]          = (_Float16)0.0f;
  h_bf[H_DIM + t]  = (_Float16)0.0f;
  h_f32[t]         = 0.0f;
  if (t == 0) *cnt = 0u;
}

// ------------------ K1: xg = X @ W_ih.T + (b_ih+b_hh) ----------------------
// Tile: 64 (s) x 32 (gate rows) per WG, 8 waves in 4x2, 16x16x32 WMMA, K=64.
__launch_bounds__(256)
__global__ void k1_xgemm(const float* __restrict__ X,
                         const float* __restrict__ Wih,
                         const float* __restrict__ b_ih,
                         const float* __restrict__ b_hh,
                         float* __restrict__ xg) {
  __shared__ __align__(16) _Float16 xs[64 * 64];   // X tile,  f16
  __shared__ __align__(16) _Float16 wt[32 * 64];   // Wih tile, f16 (row r, k)
  __shared__ float bs[32];

  const int tid = threadIdx.x;
  const int s0  = blockIdx.y * 64;
  const int r0  = blockIdx.x * 32;

  for (int idx = tid; idx < 64 * 64; idx += 256)
    xs[idx] = (_Float16)X[(size_t)(s0 + (idx >> 6)) * I_DIM + (idx & 63)];
  for (int idx = tid; idx < 32 * 64; idx += 256)
    wt[idx] = (_Float16)Wih[(size_t)(r0 + (idx >> 6)) * I_DIM + (idx & 63)];
  if (tid < 32) bs[tid] = b_ih[r0 + tid] + b_hh[r0 + tid];
  __syncthreads();

  const int w    = tid >> 5;          // wave id (wave32)
  const int lane = tid & 31;
  const int ms   = (w & 3) * 16;      // s offset of this wave's 16x16 tile
  const int ns   = (w >> 2) * 16;     // r offset
  const int lm   = lane & 15;
  const int hi16 = lane >> 4;         // 0: lanes 0-15, 1: lanes 16-31

  // two independent accumulator chains (avoid WMMA->WMMA hazard NOPs)
  v8f acc0 = {}, acc1 = {};
  {
    const int ab0 = (ms + lm) * 64 + 0  + hi16 * 8;
    const int ab1 = (ms + lm) * 64 + 32 + hi16 * 8;
    const v16h a0 = cat8(*(const v8h*)&xs[ab0], *(const v8h*)&xs[ab0 + 16]);
    const v16h a1 = cat8(*(const v8h*)&xs[ab1], *(const v8h*)&xs[ab1 + 16]);
    const int bb0 = (ns + lm) * 64 + 0  + hi16 * 8;
    const int bb1 = (ns + lm) * 64 + 32 + hi16 * 8;
    const v16h b0 = cat8(*(const v8h*)&wt[bb0], *(const v8h*)&wt[bb0 + 16]);
    const v16h b1 = cat8(*(const v8h*)&wt[bb1], *(const v8h*)&wt[bb1 + 16]);
    acc0 = __builtin_amdgcn_wmma_f32_16x16x32_f16(false, a0, false, b0,
                                                  (short)0, acc0, false, false);
    acc1 = __builtin_amdgcn_wmma_f32_16x16x32_f16(false, a1, false, b1,
                                                  (short)0, acc1, false, false);
  }
  const v8f acc = acc0 + acc1;

  // D layout: elem g -> M = g + 8*hi16, N = lm
  const int   col  = r0 + ns + lm;
  const float bias = bs[ns + lm];
  const int   srow = s0 + ms + hi16 * 8;
  #pragma unroll
  for (int g = 0; g < 8; ++g)
    xg[(size_t)(srow + g) * G_DIM + col] = acc[g] + bias;
}

// --------------------- K2: persistent LSTM recurrence ----------------------
__launch_bounds__(256, 1)
__global__ void k2_lstm(const float* __restrict__ Whh,
                        const float* __restrict__ xg,
                        _Float16* __restrict__ h_bf /*2*H doublebuf*/,
                        float* __restrict__ h_f32,
                        unsigned* __restrict__ cnt) {
  __shared__ __align__(16) _Float16 wsl[ROWS_PER_WG * H_DIM]; // 256 KB slab
  __shared__ __align__(16) _Float16 hl[H_DIM];                // 2 KB   h (f16)
  __shared__ __align__(16) float    xgl[ROWS_PER_WG];         // xg gate slices
  __shared__ float yl[ROWS_PER_WG];                           // GEMV result
  __shared__ float cl[HID_PER_WG];                            // cell state

  const int tid  = threadIdx.x;
  const int wg   = blockIdx.x;
  const int w    = tid >> 5;
  const int lane = tid & 31;
  const int lm   = lane & 15;
  const int hi16 = lane >> 4;

  // Stage W_hh slab: slab row r -> global row (r/32)*H + wg*32 + (r%32)
  for (int idx = tid; idx < ROWS_PER_WG * H_DIM; idx += 256) {
    const int r  = idx >> 10;
    const int k  = idx & 1023;
    const int gr = (r >> 5) * H_DIM + wg * HID_PER_WG + (r & 31);
    wsl[idx] = (_Float16)Whh[(size_t)gr * H_DIM + k];
  }
  if (tid < HID_PER_WG) cl[tid] = 0.0f;
  __syncthreads();

  const int      arow    = w * 16 + lm;  // this lane's slab row (A fragment)
  const unsigned hl_dst  = lds_u32(hl)  + (unsigned)tid * 8u;
  const unsigned xgl_dst = lds_u32(xgl) + (unsigned)lane * 4u;

  for (int t = 0; t < S_LEN; ++t) {
    const _Float16* hread  = h_bf + (size_t)(t & 1) * H_DIM;
    _Float16*       hwrite = h_bf + (size_t)((t + 1) & 1) * H_DIM;

    // ---- async stage h (all waves) + this WG's xg gate slices (wave 0) ----
    {
      const uint64_t hsrc = (uint64_t)(uintptr_t)(((const uint2*)hread) + tid);
      asm volatile("global_load_async_to_lds_b64 %0, %1, off"
                   :: "v"(hl_dst), "v"(hsrc) : "memory");
    }
    if (w == 0) {
      const size_t base = (size_t)t * G_DIM + wg * HID_PER_WG + lane;
      #pragma unroll
      for (int g = 0; g < 4; ++g) {
        const uint64_t src = (uint64_t)(uintptr_t)(xg + base + g * H_DIM);
        const unsigned dst = xgl_dst + (unsigned)(g * HID_PER_WG) * 4u;
        asm volatile("global_load_async_to_lds_b32 %0, %1, off"
                     :: "v"(dst), "v"(src) : "memory");
      }
      // loads complete in order: asynccnt<=4 => the h load has landed
      asm volatile("s_wait_asynccnt 0x4" ::: "memory");
    } else {
      asm volatile("s_wait_asynccnt 0x0" ::: "memory");
    }
    __syncthreads();

    // ---- y[0:128] = W_slab(128x1024) . h : 4 independent WMMA chains ----
    v8f acc0 = {}, acc1 = {}, acc2 = {}, acc3 = {};
    #pragma unroll 2
    for (int kc = 0; kc < H_DIM; kc += 128) {
      #pragma unroll
      for (int u = 0; u < 4; ++u) {
        const int k  = kc + u * 32;
        const int ab = arow * H_DIM + k + hi16 * 8;
        const v16h a = cat8(*(const v8h*)&wsl[ab], *(const v8h*)&wsl[ab + 16]);
        const int bb = k + hi16 * 8;
        const v16h b = cat8(*(const v8h*)&hl[bb], *(const v8h*)&hl[bb + 16]);
        v8f& acc = (u == 0) ? acc0 : (u == 1) ? acc1 : (u == 2) ? acc2 : acc3;
        acc = __builtin_amdgcn_wmma_f32_16x16x32_f16(false, a, false, b,
                                                     (short)0, acc, false, false);
      }
    }
    const v8f acc = (acc0 + acc1) + (acc2 + acc3);

    // extract column N=0 of D: lane 0 holds M=0..7, lane 16 holds M=8..15
    if (lane == 0) {
      #pragma unroll
      for (int g = 0; g < 8; ++g) yl[w * 16 + g] = acc[g];
    } else if (lane == 16) {
      #pragma unroll
      for (int g = 0; g < 8; ++g) yl[w * 16 + 8 + g] = acc[g];
    }
    __syncthreads();

    // ---- gate math for this WG's 32 hidden units (wave 0) ----
    if (tid < HID_PER_WG) {
      asm volatile("s_wait_asynccnt 0x0" ::: "memory");  // xgl has landed
      const int j = wg * HID_PER_WG + tid;
      float gi = yl[tid]      + xgl[tid];
      float gf = yl[32 + tid] + xgl[32 + tid];
      float gg = yl[64 + tid] + xgl[64 + tid];
      float go = yl[96 + tid] + xgl[96 + tid];
      if (t + 1 < S_LEN) {  // L2 prefetch of next step's xg slice
        const float* nx = xg + (size_t)(t + 1) * G_DIM;
        __builtin_prefetch(nx + j, 0, 1);
        __builtin_prefetch(nx + H_DIM + j, 0, 1);
        __builtin_prefetch(nx + 2 * H_DIM + j, 0, 1);
        __builtin_prefetch(nx + 3 * H_DIM + j, 0, 1);
      }
      gi = sigmoidf(gi);
      gf = sigmoidf(gf);
      gg = tanhf(gg);
      go = sigmoidf(go);
      const float c = gf * cl[tid] + gi * gg;
      cl[tid] = c;
      const float h = go * tanhf(c);
      hwrite[j] = (_Float16)h;
      h_f32[j]  = h;
      __threadfence();   // make h visible at device scope before arrival
    }
    __syncthreads();

    // ---- device-wide step barrier: monotonic counter, 1 arrival per WG ----
    if (tid == 0) {
      atomicAdd(cnt, 1u);
      const unsigned target = (unsigned)(t + 1) * NWG;
      while (__hip_atomic_load(cnt, __ATOMIC_ACQUIRE,
                               __HIP_MEMORY_SCOPE_AGENT) < target)
        __builtin_amdgcn_s_sleep(2);
    }
    __syncthreads();
  }
}

// ------------------------- K3: FC epilogue on h_final ----------------------
__launch_bounds__(256)
__global__ void k3_fc(const float* __restrict__ h,
                      const float* __restrict__ fc1w, const float* __restrict__ fc1b,
                      const float* __restrict__ fc2w, const float* __restrict__ fc2b,
                      float* __restrict__ out) {
  __shared__ float hr[H_DIM];
  __shared__ float s1[FC_DIM];
  __shared__ float red[8];
  const int tid = threadIdx.x;   // 256

  for (int k = tid; k < H_DIM; k += 256) hr[k] = fmaxf(h[k], 0.0f);
  __syncthreads();

  if (tid < FC_DIM) {
    float acc = fc1b[tid];
    const float* wr = fc1w + (size_t)tid * H_DIM;
    for (int k = 0; k < H_DIM; ++k) acc += hr[k] * wr[k];
    s1[tid] = fmaxf(acc, 0.0f);
  }
  __syncthreads();

  float p = (tid < FC_DIM) ? s1[tid] * fc2w[tid] : 0.0f;
  #pragma unroll
  for (int off = 16; off > 0; off >>= 1) p += __shfl_down(p, off, 32);
  if ((tid & 31) == 0) red[tid >> 5] = p;
  __syncthreads();
  if (tid == 0) {
    float sum = 0.0f;
    #pragma unroll
    for (int i = 0; i < 8; ++i) sum += red[i];
    out[0] = sum + fc2b[0];
  }
}

// ---------------------------------------------------------------------------
extern "C" void kernel_launch(void* const* d_in, const int* in_sizes, int n_in,
                              void* d_out, int out_size, void* d_ws, size_t ws_size,
                              hipStream_t stream) {
  const float* X    = (const float*)d_in[0];
  const float* Wih  = (const float*)d_in[1];
  const float* Whh  = (const float*)d_in[2];
  const float* bih  = (const float*)d_in[3];
  const float* bhh  = (const float*)d_in[4];
  const float* fc1w = (const float*)d_in[5];
  const float* fc1b = (const float*)d_in[6];
  const float* fc2w = (const float*)d_in[7];
  const float* fc2b = (const float*)d_in[8];
  float* out = (float*)d_out;

  // workspace layout: xg (fp32, 256 MiB) | h_bf (2*H f16) | h_f32 (H) | counter
  char*  ws  = (char*)d_ws;
  float* xg  = (float*)ws;
  size_t off = (size_t)S_LEN * G_DIM * sizeof(float);
  _Float16* h_bf = (_Float16*)(ws + off);  off += 2 * H_DIM * sizeof(_Float16);
  float*    h32  = (float*)(ws + off);     off += H_DIM * sizeof(float);
  unsigned* cnt  = (unsigned*)(ws + off);

  k0_init<<<1, 1024, 0, stream>>>(h_bf, h32, cnt);

  dim3 g1(G_DIM / 32, S_LEN / 64);
  k1_xgemm<<<g1, 256, 0, stream>>>(X, Wih, bih, bhh, xg);

  k2_lstm<<<NWG, 256, 0, stream>>>(Whh, xg, h_bf, h32, cnt);

  k3_fc<<<1, 256, 0, stream>>>(h32, fc1w, fc1b, fc2w, fc2b, out);
}
